// MHAFlash_52836687675471
// MI455X (gfx1250) — compile-verified, matrix operand order checked by reading, not run on hardware
//
#include <hip/hip_runtime.h>

#define B_  2
#define T_  2048
#define D_  1024
#define H_  16
#define DH_ 64

typedef __attribute__((ext_vector_type(16))) __bf16 v16bf;
typedef __attribute__((ext_vector_type(8)))  __bf16 v8bf;
typedef __attribute__((ext_vector_type(8)))  float  v8f;

// ---------- bf16 helpers (RNE) ----------
__device__ __forceinline__ unsigned short f32_to_bf16_bits(float f) {
    unsigned u = __builtin_bit_cast(unsigned, f);
    unsigned r = u + 0x7FFFu + ((u >> 16) & 1u);
    return (unsigned short)(r >> 16);
}
__device__ __forceinline__ float bf16_bits_to_f32(unsigned short u) {
    return __builtin_bit_cast(float, (unsigned)u << 16);
}
__device__ __forceinline__ __bf16 f32_to_bf16(float f) {
    return __builtin_bit_cast(__bf16, f32_to_bf16_bits(f));
}

// ---------- vectorized WMMA tile load ----------
// 16-bit A-matrix 16x32 tile: lane = M + 16*((K%16)/8), element e -> K =
// 16*(e/8) + 8*(lane/16) + (e%8).  Per lane: e0..7 = 8 contiguous bf16,
// e8..15 = the next 8 at +16 elements -> two 128-bit loads.
// The B-matrix 32x16 layout is the same formula with M->N, so a source that
// is row-major in the lane-walk dimension (K^T, W^T, V^T) uses this too.
__device__ __forceinline__ v16bf loadTile(const unsigned short* __restrict__ src,
                                          int ld, int row0, int k0) {
    int lane = threadIdx.x & 31, ln = lane & 15, hf = lane >> 4;
    const unsigned short* p = src + (size_t)(row0 + ln) * ld + k0 + 8 * hf;
    v8bf lo = *reinterpret_cast<const v8bf*>(p);        // global_load_b128
    v8bf hi = *reinterpret_cast<const v8bf*>(p + 16);   // global_load_b128
    return __builtin_shufflevector(lo, hi, 0,1,2,3,4,5,6,7,8,9,10,11,12,13,14,15);
}
__device__ __forceinline__ v8f wmma_bf16(v16bf a, v16bf b, v8f c) {
    return __builtin_amdgcn_wmma_f32_16x16x32_bf16(false, a, false, b, (short)0, c, false, false);
}

// ---------- prep: x (f32) -> bf16, 4 elems/thread ----------
__global__ __launch_bounds__(256) void convert_x_kernel(
    const float* __restrict__ x, unsigned short* __restrict__ xb)
{
    int i = blockIdx.x * 256 + threadIdx.x;            // [0, B*T*D/4)
    float4 v = reinterpret_cast<const float4*>(x)[i];
    unsigned lo = (unsigned)f32_to_bf16_bits(v.x) | ((unsigned)f32_to_bf16_bits(v.y) << 16);
    unsigned hi = (unsigned)f32_to_bf16_bits(v.z) | ((unsigned)f32_to_bf16_bits(v.w) << 16);
    reinterpret_cast<uint2*>(xb)[i] = make_uint2(lo, hi);
}

// ---------- prep: W (f32, [K][N]) -> Wt (bf16, [N][K]) ----------
__global__ __launch_bounds__(256) void transpose_w_kernel(
    const float* __restrict__ Wq, const float* __restrict__ Wk,
    const float* __restrict__ Wv, const float* __restrict__ Wo,
    unsigned short* __restrict__ Wqt, unsigned short* __restrict__ Wkt,
    unsigned short* __restrict__ Wvt, unsigned short* __restrict__ Wot)
{
    int z = blockIdx.z;
    const float* W = (z == 0) ? Wq : (z == 1) ? Wk : (z == 2) ? Wv : Wo;
    unsigned short* Wt = (z == 0) ? Wqt : (z == 1) ? Wkt : (z == 2) ? Wvt : Wot;
    int i = blockIdx.x * 256 + threadIdx.x;            // [0, D*D)
    int n = i & (D_ - 1), k = i >> 10;                 // coalesced read of W[k][*]
    Wt[(size_t)n * D_ + k] = f32_to_bf16_bits(W[(size_t)k * D_ + n]);
}

// ---------- Kernel 1: QKV projections, 1 wave -> 16x64 output ----------
// Q,K stored bf16 [B,H,T,DH] (pre-RoPE); V stored transposed bf16 [B,H,DH,T].
__global__ __launch_bounds__(32) void qkv_kernel(
    const unsigned short* __restrict__ xb,
    const unsigned short* __restrict__ Wqt, const float* __restrict__ bq,
    const unsigned short* __restrict__ Wkt, const float* __restrict__ bk,
    const unsigned short* __restrict__ Wvt, const float* __restrict__ bv,
    unsigned short* __restrict__ Qb, unsigned short* __restrict__ Kb,
    unsigned short* __restrict__ Vt)
{
    int rt = blockIdx.x, ct = blockIdx.y, z = blockIdx.z;
    const unsigned short* Wt = (z == 0) ? Wqt : (z == 1) ? Wkt : Wvt;
    const float* bias        = (z == 0) ? bq  : (z == 1) ? bk  : bv;
    int lane = threadIdx.x & 31, ln = lane & 15, hf = lane >> 4;

    v8f acc[4] = {{}, {}, {}, {}};
#pragma unroll 1
    for (int k0 = 0; k0 < D_; k0 += 32) {
        v16bf a = loadTile(xb, D_, rt * 16, k0);
        __builtin_prefetch(xb + (size_t)(rt * 16 + ln) * D_ + k0 + 32, 0, 1);
#pragma unroll
        for (int tt = 0; tt < 4; ++tt) {
            v16bf b = loadTile(Wt, D_, ct * 64 + tt * 16, k0);  // B[k][n] = Wt[n][k]
            acc[tt] = wmma_bf16(a, b, acc[tt]);
        }
    }
#pragma unroll
    for (int tt = 0; tt < 4; ++tt) {
        int col = ct * 64 + tt * 16 + ln;
        float bia = bias[col];
        int h = col >> 6, dh = col & 63;
#pragma unroll
        for (int v = 0; v < 8; ++v) {
            int r  = rt * 16 + v + 8 * hf;                 // row in [B*T)
            int bb = r >> 11, t = r & (T_ - 1);
            unsigned short val = f32_to_bf16_bits(acc[tt][v] + bia);
            if (z == 2)
                Vt[(((size_t)bb * H_ + h) * DH_ + dh) * T_ + t] = val;
            else if (z == 0)
                Qb[(((size_t)bb * H_ + h) * T_ + t) * DH_ + dh] = val;
            else
                Kb[(((size_t)bb * H_ + h) * T_ + t) * DH_ + dh] = val;
        }
    }
}

// ---------- Kernel 2: RoPE in place on bf16 Q,K ----------
__global__ __launch_bounds__(256) void rope_kernel(
    unsigned short* __restrict__ Qb, unsigned short* __restrict__ Kb)
{
    int idx = blockIdx.x * 256 + threadIdx.x;          // [0, B*H*T*32)
    int j  = idx & 31;
    int t  = (idx >> 5) & (T_ - 1);
    int bh = idx >> 16;
    size_t base = ((size_t)bh * T_ + t) * DH_;
    float freq = __expf(-(float)j * (9.210340371976184f / 32.0f)); // 10000^(-j/32)
    float ang = (float)t * freq;
    float s = __sinf(ang), cs = __cosf(ang);
    float q1 = bf16_bits_to_f32(Qb[base + j]);
    float q2 = bf16_bits_to_f32(Qb[base + j + 32]);
    Qb[base + j]      = f32_to_bf16_bits(q1 * cs - q2 * s);
    Qb[base + j + 32] = f32_to_bf16_bits(q1 * s + q2 * cs);
    float k1 = bf16_bits_to_f32(Kb[base + j]);
    float k2 = bf16_bits_to_f32(Kb[base + j + 32]);
    Kb[base + j]      = f32_to_bf16_bits(k1 * cs - k2 * s);
    Kb[base + j + 32] = f32_to_bf16_bits(k1 * s + k2 * cs);
}

// ---------- Kernel 3: causal flash attention, 1 wave = 16 query rows ----------
__global__ __launch_bounds__(32) void attn_kernel(
    const unsigned short* __restrict__ Qb,
    const unsigned short* __restrict__ Kb,
    const unsigned short* __restrict__ Vt,
    unsigned short* __restrict__ AOb)
{
    __shared__ __align__(16) unsigned short sPu[16][40];  // bf16 P staging
    int qt = blockIdx.x, bh = blockIdx.y;
    int lane = threadIdx.x & 31, ln = lane & 15, hf = lane >> 4;
    const unsigned short* Qh = Qb + (size_t)bh * T_ * DH_;
    const unsigned short* Kh = Kb + (size_t)bh * T_ * DH_;
    const unsigned short* Vh = Vt + (size_t)bh * T_ * DH_;   // [DH][T]

    v16bf qa0 = loadTile(Qh, DH_, qt * 16, 0);    // dh 0..31
    v16bf qa1 = loadTile(Qh, DH_, qt * 16, 32);   // dh 32..63

    v8f o0 = {}, o1 = {}, o2 = {}, o3 = {};
    float m[8], l[8];
#pragma unroll
    for (int v = 0; v < 8; ++v) { m[v] = -1e30f; l[v] = 0.0f; }

    const float scale = 0.125f;                   // 1/sqrt(64)
    int jmax = qt * 16 + 16;                      // causal bound (exclusive)
#pragma unroll 1
    for (int j0 = 0; j0 < jmax; j0 += 32) {
        // S = Q K^T : K rows gathered as the B operand (lane = key)
        v8f s0 = {}, s1 = {};
        {
            v16bf kb = loadTile(Kh, DH_, j0, 0);
            s0 = wmma_bf16(qa0, kb, s0);
            kb = loadTile(Kh, DH_, j0, 32);
            s0 = wmma_bf16(qa1, kb, s0);
            kb = loadTile(Kh, DH_, j0 + 16, 0);
            s1 = wmma_bf16(qa0, kb, s1);
            kb = loadTile(Kh, DH_, j0 + 16, 32);
            s1 = wmma_bf16(qa1, kb, s1);
        }
        int key0 = j0 + ln, key1 = j0 + 16 + ln;
#pragma unroll
        for (int v = 0; v < 8; ++v) {
            int qrow = qt * 16 + v + 8 * hf;
            float s0v = (key0 <= qrow) ? s0[v] * scale : -1e30f;
            float s1v = (key1 <= qrow) ? s1[v] * scale : -1e30f;
            float mn = fmaxf(s0v, s1v);
#pragma unroll
            for (int off = 8; off >= 1; off >>= 1)
                mn = fmaxf(mn, __shfl_xor(mn, off, 16));
            float mi = fmaxf(m[v], mn);
            float al = __expf(m[v] - mi);
            m[v] = mi;
            float p0 = __expf(s0v - mi);
            float p1 = __expf(s1v - mi);
            float rs = p0 + p1;
#pragma unroll
            for (int off = 8; off >= 1; off >>= 1)
                rs += __shfl_xor(rs, off, 16);
            l[v] = l[v] * al + rs;
            o0[v] *= al; o1[v] *= al; o2[v] *= al; o3[v] *= al;
            sPu[v + 8 * hf][ln]      = f32_to_bf16_bits(p0);
            sPu[v + 8 * hf][16 + ln] = f32_to_bf16_bits(p1);
        }
        __syncthreads();
        // reload P in A-matrix layout: two ds_load_b128 per lane
        v16bf pa;
        {
            const unsigned short* pp = &sPu[ln][8 * hf];
            v8bf lo = *reinterpret_cast<const v8bf*>(pp);
            v8bf hi = *reinterpret_cast<const v8bf*>(pp + 16);
            pa = __builtin_shufflevector(lo, hi, 0,1,2,3,4,5,6,7,8,9,10,11,12,13,14,15);
        }
        __syncthreads();
        // O += P @ V : B[k][n] = Vt[n][k] (lane walks dh rows of Vt)
        v16bf vb = loadTile(Vh, T_, 0, j0);
        o0 = wmma_bf16(pa, vb, o0);
        vb = loadTile(Vh, T_, 16, j0);
        o1 = wmma_bf16(pa, vb, o1);
        vb = loadTile(Vh, T_, 32, j0);
        o2 = wmma_bf16(pa, vb, o2);
        vb = loadTile(Vh, T_, 48, j0);
        o3 = wmma_bf16(pa, vb, o3);
    }

    int b = bh >> 4, h = bh & 15;
#pragma unroll
    for (int v = 0; v < 8; ++v) {
        float inv = 1.0f / l[v];
        int t = qt * 16 + v + 8 * hf;
        size_t rowbase = ((size_t)b * T_ + t) * D_ + h * DH_;
        AOb[rowbase + 0 * 16 + ln] = f32_to_bf16_bits(o0[v] * inv);
        AOb[rowbase + 1 * 16 + ln] = f32_to_bf16_bits(o1[v] * inv);
        AOb[rowbase + 2 * 16 + ln] = f32_to_bf16_bits(o2[v] * inv);
        AOb[rowbase + 3 * 16 + ln] = f32_to_bf16_bits(o3[v] * inv);
    }
}

// ---------- Kernel 4: output projection, 1 wave -> 16x64 output ----------
__global__ __launch_bounds__(32) void proj_kernel(
    const unsigned short* __restrict__ AOb,
    const unsigned short* __restrict__ Wot, const float* __restrict__ bo,
    float* __restrict__ out)
{
    int rt = blockIdx.x, ct = blockIdx.y;
    int lane = threadIdx.x & 31, ln = lane & 15, hf = lane >> 4;
    v8f acc[4] = {{}, {}, {}, {}};
#pragma unroll 1
    for (int k0 = 0; k0 < D_; k0 += 32) {
        v16bf a = loadTile(AOb, D_, rt * 16, k0);
        __builtin_prefetch(AOb + (size_t)(rt * 16 + ln) * D_ + k0 + 32, 0, 1);
#pragma unroll
        for (int tt = 0; tt < 4; ++tt) {
            v16bf b = loadTile(Wot, D_, ct * 64 + tt * 16, k0);
            acc[tt] = wmma_bf16(a, b, acc[tt]);
        }
    }
#pragma unroll
    for (int tt = 0; tt < 4; ++tt) {
        int col = ct * 64 + tt * 16 + ln;
        float bb = bo[col];
#pragma unroll
        for (int v = 0; v < 8; ++v) {
            int r = rt * 16 + v + 8 * hf;
            out[(size_t)r * D_ + col] = acc[tt][v] + bb;
        }
    }
}

extern "C" void kernel_launch(void* const* d_in, const int* in_sizes, int n_in,
                              void* d_out, int out_size, void* d_ws, size_t ws_size,
                              hipStream_t stream)
{
    (void)in_sizes; (void)n_in; (void)out_size; (void)ws_size;
    const float* x  = (const float*)d_in[0];
    const float* Wq = (const float*)d_in[1];
    const float* bq = (const float*)d_in[2];
    const float* Wk = (const float*)d_in[3];
    const float* bk = (const float*)d_in[4];
    const float* Wv = (const float*)d_in[5];
    const float* bv = (const float*)d_in[6];
    const float* Wo = (const float*)d_in[7];
    const float* bo = (const float*)d_in[8];
    float* out = (float*)d_out;

    const size_t NTD = (size_t)B_ * T_ * D_;     // 4,194,304 (= B*H*T*DH)
    const size_t NW  = (size_t)D_ * D_;          // 1,048,576
    unsigned short* xb  = (unsigned short*)d_ws;
    unsigned short* Wqt = xb  + NTD;
    unsigned short* Wkt = Wqt + NW;
    unsigned short* Wvt = Wkt + NW;
    unsigned short* Wot = Wvt + NW;
    unsigned short* Qb  = Wot + NW;              // [B,H,T,DH]
    unsigned short* Kb  = Qb  + NTD;
    unsigned short* Vt  = Kb  + NTD;             // [B,H,DH,T]
    unsigned short* AOb = Vt  + NTD;             // [B*T, D]

    convert_x_kernel<<<(NTD / 4) / 256, 256, 0, stream>>>(x, xb);
    transpose_w_kernel<<<dim3(NW / 256, 1, 4), 256, 0, stream>>>(
        Wq, Wk, Wv, Wo, Wqt, Wkt, Wvt, Wot);

    dim3 gA((B_ * T_) / 16, D_ / 64, 3);
    qkv_kernel<<<gA, 32, 0, stream>>>(xb, Wqt, bq, Wkt, bk, Wvt, bv, Qb, Kb, Vt);

    int ropeTotal = B_ * H_ * T_ * 32;
    rope_kernel<<<ropeTotal / 256, 256, 0, stream>>>(Qb, Kb);

    dim3 gC(T_ / 16, B_ * H_);
    attn_kernel<<<gC, 32, 0, stream>>>(Qb, Kb, Vt, AOb);

    dim3 gD((B_ * T_) / 16, D_ / 64);
    proj_kernel<<<gD, 32, 0, stream>>>(AOb, Wot, bo, out);
}
